// GAGRUCell_15547781612145
// MI455X (gfx1250) — matrix operand order repaired and checked by reading, not compiled, not run on hardware
//
#include <hip/hip_runtime.h>
#include <hip/hip_bf16.h>

// ---------------------------------------------------------------------------
// GAGRU cell for MI455X (gfx1250, wave32, WMMA).
// Attention layers: flash-style fused masked softmax using
// v_wmma_f32_16x16x32_bf16 for both the score (H H^T) and PV (A H) matmuls.
// No running-max (scores are tiny by construction; clamped at 80); the softmax
// denominator comes from a ones-column planted at padded feature 79 of the
// transposed H copy, so the matrix core computes row-sums for free.
// B-fragment streams are software-pipelined one chunk ahead so VMEM overlaps
// the matrix pipe instead of s_wait_loadcnt 0x0 before every WMMA pair.
// ---------------------------------------------------------------------------

typedef __attribute__((ext_vector_type(16))) __bf16 v16bf;
typedef __attribute__((ext_vector_type(8)))  __bf16 v8bf;
typedef __attribute__((ext_vector_type(8)))  float  v8f;

#define NN    1024
#define BB    16
#define NHEAD 3
#define HIDD  66
#define KP    96      // HID padded to multiple of 32 for K of wmma bf16
#define UNITS 64
#define NEGV  (-1e16f)
#define ONECOL 79     // padded feature index carrying the all-ones column (Ht only)

__device__ __forceinline__ __bf16 f2bf(float f) {
    unsigned u = __builtin_bit_cast(unsigned, f);
    unsigned r = u + 0x7fffu + ((u >> 16) & 1u);          // round-to-nearest-even
    return __builtin_bit_cast(__bf16, (unsigned short)(r >> 16));
}
__device__ __forceinline__ float sigm(float x) { return 1.0f / (1.0f + __expf(-x)); }
__device__ __forceinline__ v16bf pack16(v8bf lo, v8bf hi) {
    v16bf v;
#pragma unroll
    for (int t = 0; t < 8; ++t) { v[t] = lo[t]; v[t + 8] = hi[t]; }
    return v;
}

// ---------------------------------------------------------------------------
// Pack adj into a transposed bitmask: Tmask[col][word] bit r = (adj[r][col]!=0)
// ---------------------------------------------------------------------------
__global__ void maskprep(const int* __restrict__ adj, unsigned* __restrict__ Tm) {
    int idx = blockIdx.x * blockDim.x + threadIdx.x;      // total N*32
    if (idx >= NN * 32) return;
    int w = idx & 31, c = idx >> 5;
    unsigned word = 0u;
    for (int r = 0; r < 32; ++r)
        word |= (adj[(size_t)(w * 32 + r) * NN + c] != 0 ? 1u : 0u) << r;
    Tm[(size_t)c * 32 + w] = word;
}

// ---------------------------------------------------------------------------
// X1 = concat(x, h); also pre-write the x-part of X2 (h-part filled by ru_kernel)
// ---------------------------------------------------------------------------
__global__ void concat_x(const float* __restrict__ x, const float* __restrict__ hx,
                         float* __restrict__ X1, float* __restrict__ X2) {
    size_t idx = (size_t)blockIdx.x * blockDim.x + threadIdx.x; // B*N*66
    if (idx >= (size_t)BB * NN * HIDD) return;
    int k = (int)(idx % HIDD);
    size_t r = idx / HIDD;                                 // b*N + n
    if (k < 2) { float v = x[r * 2 + k]; X1[idx] = v; X2[idx] = v; }
    else        X1[idx] = hx[r * UNITS + (k - 2)];
}

// ---------------------------------------------------------------------------
// Per-head projection: H = X @ Wh[head] + bh[head], bf16, zero-padded to KP.
// Transposed copy gets the ones-column at feature ONECOL (PV row-sum trick).
// ---------------------------------------------------------------------------
__global__ void proj_heads(const float* __restrict__ X, const float* __restrict__ Wh,
                           const float* __restrict__ bh,
                           __bf16* __restrict__ Hb, __bf16* __restrict__ HbT) {
    size_t idx = (size_t)blockIdx.x * blockDim.x + threadIdx.x; // 3*B*N*KP
    if (idx >= (size_t)NHEAD * BB * NN * KP) return;
    int e = (int)(idx % KP);
    size_t row = idx / KP;
    int n = (int)(row % NN);
    int beff = (int)(row / NN);                            // head*B + b
    int b = beff & (BB - 1), head = beff >> 4;
    float acc = 0.f;
    if (e < HIDD) {
        acc = bh[head * HIDD + e];
        const float* xr = X + ((size_t)b * NN + n) * HIDD;
        const float* w  = Wh + (size_t)head * HIDD * HIDD + e;
        for (int k = 0; k < HIDD; ++k) acc += xr[k] * w[(size_t)k * HIDD];
    }
    __bf16 v = f2bf(acc);
    Hb[idx] = v;
    HbT[((size_t)beff * KP + e) * NN + n] = (e == ONECOL) ? f2bf(1.0f) : v;
}

// ---------------------------------------------------------------------------
// Output-layer projection: Hout = lrelu'd-concat(heads) @ Wo + bo.
// Hatt already has leaky_relu applied by the attention epilogue.
// ---------------------------------------------------------------------------
__global__ void proj_out(const float* __restrict__ Hatt, const float* __restrict__ Wo,
                         const float* __restrict__ bo,
                         __bf16* __restrict__ Hb, __bf16* __restrict__ HbT) {
    size_t idx = (size_t)blockIdx.x * blockDim.x + threadIdx.x; // B*N*KP
    if (idx >= (size_t)BB * NN * KP) return;
    int e = (int)(idx % KP);
    size_t row = idx / KP;
    int n = (int)(row % NN);
    int b = (int)(row / NN);
    float acc = 0.f;
    if (e < HIDD) {
        acc = bo[e];
        for (int head = 0; head < NHEAD; ++head) {
            const float* hr = Hatt + (((size_t)(head * BB + b)) * NN + n) * HIDD;
            const float* w  = Wo + (size_t)head * HIDD * HIDD + e;
            for (int k = 0; k < HIDD; ++k) acc += hr[k] * w[(size_t)k * HIDD];
        }
    }
    __bf16 v = f2bf(acc);
    Hb[idx] = v;
    HbT[((size_t)b * KP + e) * NN + n] = (e == ONECOL) ? f2bf(1.0f) : v;
}

// ---------------------------------------------------------------------------
// Fused masked-softmax attention. One wave owns TWO 16-row tiles (32 rows):
// score/PV B-fragments and the mask word are loaded once and reused by both.
// Out = leaky_relu( softmax(mask(H H^T)) @ H ), f32, stride HIDD.
// Grid must be Beff*8 blocks of 128 threads (4 waves): EXEC always all-ones.
// ---------------------------------------------------------------------------
__global__ __launch_bounds__(128) void attn_kernel(
    const __bf16* __restrict__ Hrow,   // [Beff][NN][KP]
    const __bf16* __restrict__ Ht,     // [Beff][KP][NN] (+ones at row ONECOL)
    const unsigned* __restrict__ Tm,   // [NN][NN/32]
    float* __restrict__ Out)           // [Beff][NN][HIDD]
{
    __shared__ __align__(16) __bf16 plds[4][256];          // 16x16 bf16 per wave
    const int lane  = threadIdx.x & 31;
    const int wave  = threadIdx.x >> 5;
    const int gidx  = blockIdx.x * 4 + wave;
    const int beff  = gidx >> 3;
    const int rpair = gidx & 7;                            // rows rpair*32..+31
    const int half  = lane >> 4;
    const int l16   = lane & 15;

    // A fragments (contiguous k = 16*half + s, same order as B), two row tiles, K padded to 96.
    v16bf a[2][3];
#pragma unroll
    for (int r = 0; r < 2; ++r) {
        const __bf16* arow =
            Hrow + ((size_t)beff * NN + rpair * 32 + r * 16 + l16) * KP;
#pragma unroll
        for (int c = 0; c < 3; ++c) {
            v8bf lo = *(const v8bf*)(arow + 32 * c + half * 16);
            v8bf hi = *(const v8bf*)(arow + 32 * c + 8 + half * 16);
            a[r][c] = pack16(lo, hi);
        }
    }

    v8f o[2][5] = {};                                      // d padded 66->80; d=79 = row-sum

    // Streaming pointers, advanced by 16 columns per iteration.
    const __bf16*   brow = Hrow + ((size_t)beff * NN + l16) * KP + half * 16;
    const __bf16*   btj  = Ht + (size_t)beff * KP * NN;
    const unsigned* tmp  = Tm + (size_t)l16 * (NN / 32) + rpair;

    for (int j = 0; j < NN / 16; ++j) {
        // mask word in flight during the score matmuls
        const unsigned mw = *tmp;

        // ---- S tiles, 1-deep pipelined B-fragment stream (shared by both A sets)
        v8bf lo0 = *(const v8bf*)(brow);
        v8bf hi0 = *(const v8bf*)(brow + 8);
        v8f s0 = {}, s1 = {};
        {
            v8bf lo1 = *(const v8bf*)(brow + 32);
            v8bf hi1 = *(const v8bf*)(brow + 40);
            v16bf bf = pack16(lo0, hi0);
            s0 = __builtin_amdgcn_wmma_f32_16x16x32_bf16(
                     false, a[0][0], false, bf, (short)0, s0, false, false);
            s1 = __builtin_amdgcn_wmma_f32_16x16x32_bf16(
                     false, a[1][0], false, bf, (short)0, s1, false, false);
            lo0 = *(const v8bf*)(brow + 64);
            hi0 = *(const v8bf*)(brow + 72);
            bf = pack16(lo1, hi1);
            s0 = __builtin_amdgcn_wmma_f32_16x16x32_bf16(
                     false, a[0][1], false, bf, (short)0, s0, false, false);
            s1 = __builtin_amdgcn_wmma_f32_16x16x32_bf16(
                     false, a[1][1], false, bf, (short)0, s1, false, false);
            bf = pack16(lo0, hi0);
            s0 = __builtin_amdgcn_wmma_f32_16x16x32_bf16(
                     false, a[0][2], false, bf, (short)0, s0, false, false);
            s1 = __builtin_amdgcn_wmma_f32_16x16x32_bf16(
                     false, a[1][2], false, bf, (short)0, s1, false, false);
        }

        // ---- mask + exp (no running max; scores are tiny, clamp for safety) ----
        v16bf pa[2];
#pragma unroll
        for (int r = 0; r < 2; ++r) {
            const int msh = (r << 4) + half * 8;           // bit = row - rpair*32
#pragma unroll
            for (int i = 0; i < 8; ++i) {
                float sv = ((mw >> (msh + i)) & 1u) ? (r ? s1[i] : s0[i]) : NEGV;
                float p  = __expf(fminf(sv, 80.f));        // masked -> exp(-1e16)=0
                plds[wave][(half * 8 + i) * 16 + l16] = f2bf(p); // C-layout -> LDS
            }
            asm volatile("" ::: "memory");                 // keep DS store->load order
            v8bf lo = *(const v8bf*)(&plds[wave][l16 * 16 + half * 8]);
#pragma unroll
            for (int t = 0; t < 8; ++t) {                  // P in A-layout, K 16..31 = 0
                pa[r][t]     = lo[t];
                pa[r][t + 8] = __builtin_bit_cast(__bf16, (unsigned short)0);
            }
            asm volatile("" ::: "memory");
        }

        // ---- O += P x H_cols; pipelined B stream from transposed copy ----
        {
            const __bf16* bp = btj + (size_t)l16 * NN;
            v8bf plo = *(const v8bf*)(bp + half * 8);
            v8bf phi = *(const v8bf*)(bp + 8);
#pragma unroll
            for (int nc = 0; nc < 5; ++nc) {
                v8bf nlo, nhi;
                if (nc < 4) {
                    const __bf16* np = btj + (size_t)((nc + 1) * 16 + l16) * NN;
                    nlo = *(const v8bf*)(np + half * 8);
                    nhi = *(const v8bf*)(np + 8);
                }
                v16bf bf = pack16(plo, phi);
                o[0][nc] = __builtin_amdgcn_wmma_f32_16x16x32_bf16(
                               false, pa[0], false, bf, (short)0, o[0][nc], false, false);
                o[1][nc] = __builtin_amdgcn_wmma_f32_16x16x32_bf16(
                               false, pa[1], false, bf, (short)0, o[1][nc], false, false);
                plo = nlo; phi = nhi;
            }
        }

        brow += (size_t)16 * KP;
        btj  += 16;
        tmp  += (size_t)16 * (NN / 32);
    }

    // ---- normalize by row-sum (broadcast from lane holding d=79), lrelu, store
#pragma unroll
    for (int r = 0; r < 2; ++r) {
#pragma unroll
        for (int i = 0; i < 8; ++i) {
            const float lsum = __shfl(o[r][4][i], (lane & 16) | 15, 32);
            const float inv  = 1.0f / lsum;
            float* op = Out + ((size_t)beff * NN + rpair * 32 + r * 16
                               + half * 8 + i) * HIDD;
#pragma unroll
            for (int nc = 0; nc < 5; ++nc) {
                const int d = nc * 16 + l16;
                if (d < HIDD) {
                    float v = o[r][nc][i] * inv;
                    op[d] = v > 0.f ? v : 0.01f * v;
                }
            }
        }
    }
}

// ---------------------------------------------------------------------------
// value = sigmoid(sigmoid(G1 @ W_ru) + b_ru); r -> X2 h-part (r*h), u -> U
// ---------------------------------------------------------------------------
__global__ void ru_kernel(const float* __restrict__ G1, const float* __restrict__ Wru,
                          const float* __restrict__ bru, const float* __restrict__ hx,
                          float* __restrict__ X2, float* __restrict__ U) {
    size_t idx = (size_t)blockIdx.x * blockDim.x + threadIdx.x; // B*N*128
    if (idx >= (size_t)BB * NN * 128) return;
    int j = (int)(idx & 127);
    size_t r = idx >> 7;
    const float* g = G1 + r * HIDD;
    float acc = 0.f;
    for (int k = 0; k < HIDD; ++k) acc += g[k] * Wru[(size_t)k * 128 + j];
    float v = sigm(sigm(acc) + bru[j]);
    if (j < UNITS) X2[r * HIDD + 2 + j] = v * hx[r * UNITS + j];
    else           U[r * UNITS + (j - UNITS)] = v;
}

// ---------------------------------------------------------------------------
// c = tanh(sigmoid(G2 @ W_c) + b_c); out = u*h + (1-u)*c
// ---------------------------------------------------------------------------
__global__ void final_kernel(const float* __restrict__ G2, const float* __restrict__ Wc,
                             const float* __restrict__ bc, const float* __restrict__ U,
                             const float* __restrict__ hx, float* __restrict__ out) {
    size_t idx = (size_t)blockIdx.x * blockDim.x + threadIdx.x; // B*N*64
    if (idx >= (size_t)BB * NN * UNITS) return;
    int j = (int)(idx & 63);
    size_t r = idx >> 6;
    const float* g = G2 + r * HIDD;
    float acc = 0.f;
    for (int k = 0; k < HIDD; ++k) acc += g[k] * Wc[(size_t)k * UNITS + j];
    float c = tanhf(sigm(acc) + bc[j]);
    float u = U[idx], h = hx[idx];
    out[idx] = u * h + (1.f - u) * c;
}

// ---------------------------------------------------------------------------
extern "C" void kernel_launch(void* const* d_in, const int* in_sizes, int n_in,
                              void* d_out, int out_size, void* d_ws, size_t ws_size,
                              hipStream_t stream) {
    const float* x   = (const float*)d_in[0];
    const float* hx  = (const float*)d_in[1];
    const int*   adj = (const int*)d_in[2];
    const float* Wh  = (const float*)d_in[3];
    const float* bh  = (const float*)d_in[4];
    const float* Wo  = (const float*)d_in[5];
    const float* bo  = (const float*)d_in[6];
    const float* Wru = (const float*)d_in[7];
    const float* bru = (const float*)d_in[8];
    const float* Wc  = (const float*)d_in[9];
    const float* bc  = (const float*)d_in[10];
    float* out = (float*)d_out;

    char* ws = (char*)d_ws;                                // ~57 MB used
    float*    X1   = (float*)   (ws + 0);                  //  4,325,376
    float*    X2   = (float*)   (ws + 4325376);            //  4,325,376
    __bf16*   Hbf  = (__bf16*)  (ws + 8650752);            //  9,437,184
    __bf16*   HbfT = (__bf16*)  (ws + 18087936);           //  9,437,184
    float*    Hatt = (float*)   (ws + 27525120);           // 12,976,128
    __bf16*   Obf  = (__bf16*)  (ws + 40501248);           //  3,145,728
    __bf16*   ObfT = (__bf16*)  (ws + 43646976);           //  3,145,728
    float*    G1   = (float*)   (ws + 46792704);           //  4,325,376
    float*    G2   = (float*)   (ws + 51118080);           //  4,325,376
    float*    U    = (float*)   (ws + 55443456);           //  4,194,304
    unsigned* Tm   = (unsigned*)(ws + 59637760);           //    131,072

    maskprep  <<<128,   256, 0, stream>>>(adj, Tm);
    concat_x  <<<4224,  256, 0, stream>>>(x, hx, X1, X2);

    // ---- subnet 1: g_ru ----
    proj_heads<<<18432, 256, 0, stream>>>(X1, Wh, bh, Hbf, HbfT);
    attn_kernel<<<NHEAD * BB * (NN / 128), 128, 0, stream>>>(Hbf, HbfT, Tm, Hatt);
    proj_out  <<<6144,  256, 0, stream>>>(Hatt, Wo, bo, Obf, ObfT);
    attn_kernel<<<BB * (NN / 128), 128, 0, stream>>>(Obf, ObfT, Tm, G1);
    ru_kernel <<<8192,  256, 0, stream>>>(G1, Wru, bru, hx, X2, U);

    // ---- subnet 2: g_c ----
    proj_heads<<<18432, 256, 0, stream>>>(X2, Wh, bh, Hbf, HbfT);
    attn_kernel<<<NHEAD * BB * (NN / 128), 128, 0, stream>>>(Hbf, HbfT, Tm, Hatt);
    proj_out  <<<6144,  256, 0, stream>>>(Hatt, Wo, bo, Obf, ObfT);
    attn_kernel<<<BB * (NN / 128), 128, 0, stream>>>(Obf, ObfT, Tm, G2);

    final_kernel<<<4096, 256, 0, stream>>>(G2, Wc, bc, U, hx, out);
}
